// Attention_86655260164582
// MI455X (gfx1250) — matrix operand order repaired and checked by reading, not compiled, not run on hardware
//
#include <hip/hip_runtime.h>
#include <hip/hip_bf16.h>

// ---------------------------------------------------------------------------
// Problem constants (match reference)
// ---------------------------------------------------------------------------
constexpr int kT   = 2048;   // sequence length
constexpr int kC   = 1024;   // model dim
constexpr int kH   = 16;     // query heads
constexpr int kKVH = 8;      // kv heads
constexpr int kHD  = 64;     // head dim
constexpr int kVE  = 12;     // gate input channels

typedef __attribute__((ext_vector_type(16))) __bf16 v16bf;
typedef __attribute__((ext_vector_type(8)))  float  v8f;

// ---------------------------------------------------------------------------
// fp32 -> bf16 convert (grid-stride)
// ---------------------------------------------------------------------------
__global__ void f32_to_bf16_kernel(const float* __restrict__ in,
                                   __bf16* __restrict__ out, size_t n) {
  size_t i = (size_t)blockIdx.x * blockDim.x + threadIdx.x;
  size_t stride = (size_t)gridDim.x * blockDim.x;
  for (; i < n; i += stride) out[i] = (__bf16)in[i];
}

// ---------------------------------------------------------------------------
// fp32 (K x N row-major) -> bf16 transposed (N x K row-major), LDS-tiled.
// K, N multiples of 32. 256 threads = 32x8 tile workers.
// Makes WMMA B-fragment loads K-contiguous per lane (global_load_b128).
// ---------------------------------------------------------------------------
__global__ void transpose_f32_to_bf16_kernel(const float* __restrict__ in,
                                             __bf16* __restrict__ out,
                                             int K, int N) {
  __shared__ float tile[32][33];   // +1 pad: no bank conflicts
  const int kb = blockIdx.x * 32;
  const int nb = blockIdx.y * 32;
  const int tx = threadIdx.x & 31;
  const int ty = threadIdx.x >> 5;
  for (int r = ty; r < 32; r += 8)
    tile[r][tx] = in[(size_t)(kb + r) * N + nb + tx];
  __syncthreads();
  for (int r = ty; r < 32; r += 8)
    out[(size_t)(nb + r) * K + kb + tx] = (__bf16)tile[tx][r];
}

// ---------------------------------------------------------------------------
// bf16 GEMM via v_wmma_f32_16x16x32_bf16 with PRE-TRANSPOSED B.
// A:  M x K row-major bf16.
// Bt: N x K row-major bf16 (i.e. original K x N matrix transposed).
// C:  fp32, ldc/coff select a column window. One wave = 16x64 tile.
// A fragment (16x32): lanes 0-15 row=lane, elems K[0..15];
//                     lanes 16-31 row=lane-16, elems K[16..31].
// B fragment (32x16): lanes 0-15 col=lane, elems K[0..15];
//                     lanes 16-31 col=lane-16, elems K[16..31].
// Both are 16-halfword contiguous runs -> 2x global_load_b128 each.
// ---------------------------------------------------------------------------
__global__ void gemm_bf16_wmma_kernel(const __bf16* __restrict__ A,
                                      const __bf16* __restrict__ Bt,
                                      float* __restrict__ Cout,
                                      int M, int N, int K, int ldc, int coff) {
  const int lane = threadIdx.x & 31;
  const int wid  = threadIdx.x >> 5;
  const int tilesN = N >> 6;               // 64 cols per wave tile
  const int tilesM = M >> 4;               // 16 rows per wave tile
  int id = blockIdx.x * (blockDim.x >> 5) + wid;
  if (id >= tilesM * tilesN) return;
  const int tm = id / tilesN;
  const int tn = id % tilesN;

  const int l15   = lane & 15;
  const int koff  = (lane < 16) ? 0 : 16;  // A element K offset
  const int krow0 = (lane < 16) ? 0 : 16;  // B element K offset

  v8f acc[4] = {};
  const __bf16* aBase = A + (size_t)(tm * 16 + l15) * K;
  const __bf16* bBase0 = Bt + (size_t)(tn * 64 + l15) * K + krow0;

  for (int kk = 0; kk < K; kk += 32) {
    const __bf16* ap = aBase + kk;
    __builtin_prefetch(ap + 128, 0, 1);    // global_prefetch_b8 ahead on A
    v16bf a;
#pragma unroll
    for (int e = 0; e < 8; ++e) a[e] = ap[koff + e];
#pragma unroll
    for (int e = 0; e < 8; ++e) a[8 + e] = ap[8 + koff + e];

#pragma unroll
    for (int j = 0; j < 4; ++j) {
      const __bf16* bp = bBase0 + (size_t)j * 16 * K + kk;  // contiguous 16 halfs
      v16bf b;
#pragma unroll
      for (int e = 0; e < 16; ++e) b[e] = bp[e];
      acc[j] = __builtin_amdgcn_wmma_f32_16x16x32_bf16(
          false, a, false, b, (short)0, acc[j], false, false);
    }
  }

  // D layout: VGPR r -> M = r (+8 for lanes 16-31), N = lane&15
  const int mbase = tm * 16 + ((lane < 16) ? 0 : 8);
#pragma unroll
  for (int j = 0; j < 4; ++j)
#pragma unroll
    for (int r = 0; r < 8; ++r)
      Cout[(size_t)(mbase + r) * ldc + coff + tn * 64 + j * 16 + l15] = acc[j][r];
}

// ---------------------------------------------------------------------------
// Epilogue: gate + ve add, RoPE, RMS-norm * 1.2; emit bf16 head-major q/k
// plus a key-major transposed v for the w@v WMMA B operand.
// One block per token t; 8 waves: wave w -> q heads {2w,2w+1}, kv head w.
// qkv fp32 layout: row t holds [q(0..1023) | k(1024..1535) | v(1536..2047)].
// ---------------------------------------------------------------------------
__global__ void qkv_epilogue_kernel(const float* __restrict__ qkv,
                                    const float* __restrict__ x,
                                    const float* __restrict__ ve,
                                    const float* __restrict__ cosb,
                                    const float* __restrict__ sinb,
                                    const float* __restrict__ Wg,
                                    __bf16* __restrict__ qb,   // (H, T, HD)
                                    __bf16* __restrict__ kb,   // (KVH, T, HD)
                                    __bf16* __restrict__ vt) { // (KVH, HD, T)
  const int t    = blockIdx.x;
  const int lane = threadIdx.x & 31;
  const int wid  = threadIdx.x >> 5;
  const float c = cosb[t * (kHD / 2) + lane];
  const float s = sinb[t * (kHD / 2) + lane];
  const float* row = qkv + (size_t)t * 2048;

  // ---- q heads (rope + rms) ----
#pragma unroll
  for (int hh = 0; hh < 2; ++hh) {
    const int h = wid * 2 + hh;
    float x1 = row[h * kHD + lane];
    float x2 = row[h * kHD + 32 + lane];
    float r1 = x1 * c + x2 * s;
    float r2 = -x1 * s + x2 * c;
    float ss = r1 * r1 + r2 * r2;
#pragma unroll
    for (int o = 16; o; o >>= 1) ss += __shfl_xor(ss, o, 32);
    float inv = rsqrtf(ss * (1.0f / kHD) + 1e-6f) * 1.2f;
    __bf16* qp = qb + ((size_t)h * kT + t) * kHD;
    qp[lane]      = (__bf16)(r1 * inv);
    qp[lane + 32] = (__bf16)(r2 * inv);
  }

  // ---- kv head (k: rope+rms, v: gate*ve add, no norm) ----
  {
    const int kh = wid;
    float x1 = row[1024 + kh * kHD + lane];
    float x2 = row[1024 + kh * kHD + 32 + lane];
    float r1 = x1 * c + x2 * s;
    float r2 = -x1 * s + x2 * c;
    float ss = r1 * r1 + r2 * r2;
#pragma unroll
    for (int o = 16; o; o >>= 1) ss += __shfl_xor(ss, o, 32);
    float inv = rsqrtf(ss * (1.0f / kHD) + 1e-6f) * 1.2f;
    __bf16* kp = kb + ((size_t)kh * kT + t) * kHD;
    kp[lane]      = (__bf16)(r1 * inv);
    kp[lane + 32] = (__bf16)(r2 * inv);

    float z = 0.f;
#pragma unroll
    for (int i2 = 0; i2 < kVE; ++i2) z += x[(size_t)t * kC + i2] * Wg[i2 * kKVH + kh];
    float g = 3.0f / (1.0f + __expf(-z));
    float v1 = row[1536 + kh * kHD + lane]      + g * ve[(size_t)t * (kKVH * kHD) + kh * kHD + lane];
    float v2 = row[1536 + kh * kHD + 32 + lane] + g * ve[(size_t)t * (kKVH * kHD) + kh * kHD + 32 + lane];
    // key-major transposed v: vt[kvh][d][t]
    vt[((size_t)kh * kHD + lane)      * kT + t] = (__bf16)v1;
    vt[((size_t)kh * kHD + lane + 32) * kT + t] = (__bf16)v2;
  }
}

// ---------------------------------------------------------------------------
// Fused attention: one block per (query-tile of 16, head). 8 waves, 256 thr.
// Dynamic LDS: 16 x 2048 fp32 scores (128 KB; CDNA5 WGP has 320 KB).
// Phase 1: scores = (q k^T)/8 via WMMA -> LDS
// Phase 2: += ca_bias + alpha * depthwise3x3(prev_attn), causal only
// Phase 3: row softmax (wave32 shuffle reductions), NT-write w, zero tail
// Phase 4: y = w @ v via WMMA (split-K over 2 wave groups, LDS reduce)
// ---------------------------------------------------------------------------
__global__ void attention_kernel(const __bf16* __restrict__ qb,
                                 const __bf16* __restrict__ kb,
                                 const __bf16* __restrict__ vt,    // (KVH,HD,T)
                                 const float* __restrict__ prev,   // (H,T,T)
                                 const float* __restrict__ cbias,  // (T,T)
                                 const float* __restrict__ convw,  // (H,9)
                                 const float* __restrict__ alphap, // scalar
                                 float* __restrict__ wout,         // (H,T,T)
                                 __bf16* __restrict__ yb) {        // (T, H*HD)
  extern __shared__ float sc[];  // [16][kT]
  const int qt  = blockIdx.x;    // query tile
  const int h   = blockIdx.y;    // head
  const int kvh = h >> 1;        // GQA: 2 q-heads per kv-head
  const int tid  = threadIdx.x;
  const int lane = tid & 31;
  const int wid  = tid >> 5;
  const int l15   = lane & 15;
  const int koff  = (lane < 16) ? 0 : 16;
  const int krow0 = (lane < 16) ? 0 : 16;
  const int mhi   = (lane < 16) ? 0 : 8;

  // ---------------- Phase 1: q k^T ----------------
  {
    const __bf16* qrow = qb + ((size_t)h * kT + qt * 16 + l15) * kHD;
    for (int kt = wid; kt <= qt; kt += 8) {
      v8f acc = {};
#pragma unroll
      for (int kc = 0; kc < 2; ++kc) {
        const int kk = kc * 32;
        v16bf a;
#pragma unroll
        for (int e = 0; e < 8; ++e) a[e] = qrow[kk + koff + e];
#pragma unroll
        for (int e = 0; e < 8; ++e) a[8 + e] = qrow[kk + 8 + koff + e];
        // B[kd][n] = k[key = kt*16+n][d = kk+krow0+e]  (d-contiguous per lane)
        const __bf16* kp = kb + ((size_t)kvh * kT + kt * 16 + l15) * kHD + kk + krow0;
        v16bf b;
#pragma unroll
        for (int e = 0; e < 16; ++e) b[e] = kp[e];
        acc = __builtin_amdgcn_wmma_f32_16x16x32_bf16(
            false, a, false, b, (short)0, acc, false, false);
      }
#pragma unroll
      for (int r = 0; r < 8; ++r)
        sc[(r + mhi) * kT + kt * 16 + l15] = acc[r] * 0.125f;  // 1/sqrt(64)
    }
  }
  __syncthreads();

  // ---------------- Phase 2: bias + alpha * conv(prev_attn) ----------------
  {
    float cw[9];
#pragma unroll
    for (int i = 0; i < 9; ++i) cw[i] = convw[h * 9 + i];
    const float al = alphap[0];
    const float* ph = prev + (size_t)h * kT * kT;
    for (int m = 0; m < 16; ++m) {
      const int i = qt * 16 + m;
      for (int j = tid; j <= i; j += 256) {
        float r = 0.f;
#pragma unroll
        for (int di = 0; di < 3; ++di) {
          const int ii = i + di - 1;
          if (ii < 0 || ii >= kT) continue;
#pragma unroll
          for (int dj = 0; dj < 3; ++dj) {
            const int jj = j + dj - 1;
            if (jj < 0 || jj >= kT) continue;
            r += cw[di * 3 + dj] * ph[(size_t)ii * kT + jj];
          }
        }
        sc[m * kT + j] += cbias[(size_t)i * kT + j] + al * r;
      }
    }
  }
  __syncthreads();

  // ---------------- Phase 3: row softmax, emit w (NT), zero causal tail ----
  {
#pragma unroll
    for (int mm = 0; mm < 2; ++mm) {
      const int m = wid * 2 + mm;
      const int i = qt * 16 + m;
      const int limit = i + 1;
      float mx = -1e30f;
      for (int j = lane; j < limit; j += 32) mx = fmaxf(mx, sc[m * kT + j]);
#pragma unroll
      for (int o = 16; o; o >>= 1) mx = fmaxf(mx, __shfl_xor(mx, o, 32));
      float sum = 0.f;
      for (int j = lane; j < limit; j += 32) {
        float e = __expf(sc[m * kT + j] - mx);
        sc[m * kT + j] = e;
        sum += e;
      }
#pragma unroll
      for (int o = 16; o; o >>= 1) sum += __shfl_xor(sum, o, 32);
      const float inv = 1.0f / sum;
      float* wrow = wout + ((size_t)h * kT + i) * kT;
      for (int j = lane; j < limit; j += 32) {
        float v = sc[m * kT + j] * inv;
        sc[m * kT + j] = v;
        __builtin_nontemporal_store(v, &wrow[j]);   // w never re-read: NT
      }
      for (int j = limit + lane; j < kT; j += 32) {
        sc[m * kT + j] = 0.f;
        __builtin_nontemporal_store(0.f, &wrow[j]);
      }
    }
  }
  __syncthreads();

  // ---------------- Phase 4: y = w @ v (split-K, WMMA) ----------------
  {
    const int col   = wid & 3;       // 16-wide d-column of the 16x64 y tile
    const int split = wid >> 2;      // split-K group
    const int nch   = (qt + 2) >> 1; // ceil((qt+1)*16 / 32) key chunks
    v8f acc = {};
    const __bf16* vBase = vt + ((size_t)kvh * kHD + col * 16 + l15) * kT + krow0;
    for (int c = split; c < nch; c += 2) {
      const int kk = c * 32;
      v16bf a;  // w tile 16x32, re-quantized bf16 from LDS
#pragma unroll
      for (int e = 0; e < 8; ++e) a[e] = (__bf16)sc[l15 * kT + kk + koff + e];
#pragma unroll
      for (int e = 0; e < 8; ++e) a[8 + e] = (__bf16)sc[l15 * kT + kk + 8 + koff + e];
      // B[key][d] from key-major vt: contiguous 16 halfs per lane
      const __bf16* vp = vBase + kk;
      v16bf b;
#pragma unroll
      for (int e = 0; e < 16; ++e) b[e] = vp[e];
      acc = __builtin_amdgcn_wmma_f32_16x16x32_bf16(
          false, a, false, b, (short)0, acc, false, false);
    }
    __syncthreads();  // all LDS score reads complete -> safe to reuse as scratch
    float* red = sc;  // 8 waves x 8 regs x 32 lanes = 8 KB
#pragma unroll
    for (int r = 0; r < 8; ++r) red[(wid * 8 + r) * 32 + lane] = acc[r];
    __syncthreads();
    if (wid < 4) {
#pragma unroll
      for (int r = 0; r < 8; ++r) {
        float v = red[(wid * 8 + r) * 32 + lane] + red[((wid + 4) * 8 + r) * 32 + lane];
        const int m = r + mhi;
        yb[((size_t)(qt * 16 + m)) * kC + h * kHD + wid * 16 + l15] = (__bf16)v;
      }
    }
  }
}

// ---------------------------------------------------------------------------
// Launcher
// ---------------------------------------------------------------------------
extern "C" void kernel_launch(void* const* d_in, const int* in_sizes, int n_in,
                              void* d_out, int out_size, void* d_ws, size_t ws_size,
                              hipStream_t stream) {
  const float* x     = (const float*)d_in[0];
  const float* ve    = (const float*)d_in[1];
  const float* cosb  = (const float*)d_in[2];
  const float* sinb  = (const float*)d_in[3];
  const float* prev  = (const float*)d_in[4];
  const float* Wq    = (const float*)d_in[5];
  const float* Wk    = (const float*)d_in[6];
  const float* Wv    = (const float*)d_in[7];
  const float* Wo    = (const float*)d_in[8];
  const float* Wg    = (const float*)d_in[9];
  const float* cbias = (const float*)d_in[10];
  const float* convw = (const float*)d_in[11];
  const float* alpha = (const float*)d_in[12];

  float* outp = (float*)d_out;                      // (T, C) fp32
  float* wout = outp + (size_t)kT * kC;             // (H, T, T) fp32

  // workspace layout (bytes)
  char* ws = (char*)d_ws;
  __bf16* xb  = (__bf16*)(ws + (size_t)0);           // 4 MB  x bf16
  __bf16* wqt = (__bf16*)(ws + ((size_t)4  << 20));  // 2 MB  Wq^T bf16 (N x K)
  __bf16* wkt = (__bf16*)(ws + ((size_t)6  << 20));  // 1 MB  Wk^T
  __bf16* wvt = (__bf16*)(ws + ((size_t)7  << 20));  // 1 MB  Wv^T
  __bf16* wot = (__bf16*)(ws + ((size_t)8  << 20));  // 2 MB  Wo^T
  float*  qkv = (float*) (ws + ((size_t)10 << 20));  // 16 MB (T x 2048 fp32)
  __bf16* qb  = (__bf16*)(ws + ((size_t)26 << 20));  // 4 MB (H,T,HD)
  __bf16* kb  = (__bf16*)(ws + ((size_t)30 << 20));  // 2 MB (KVH,T,HD)
  __bf16* vt  = (__bf16*)(ws + ((size_t)32 << 20));  // 2 MB (KVH,HD,T)
  __bf16* yb  = (__bf16*)(ws + ((size_t)34 << 20));  // 4 MB (T, C)

  // 1) fp32 -> bf16 conversions (weights transposed for K-contiguous B frags)
  f32_to_bf16_kernel<<<2048, 256, 0, stream>>>(x, xb, (size_t)kT * kC);
  transpose_f32_to_bf16_kernel<<<dim3(kC / 32, 1024 / 32), 256, 0, stream>>>(Wq, wqt, kC, 1024);
  transpose_f32_to_bf16_kernel<<<dim3(kC / 32, 512 / 32),  256, 0, stream>>>(Wk, wkt, kC, 512);
  transpose_f32_to_bf16_kernel<<<dim3(kC / 32, 512 / 32),  256, 0, stream>>>(Wv, wvt, kC, 512);
  transpose_f32_to_bf16_kernel<<<dim3(kC / 32, kC / 32),   256, 0, stream>>>(Wo, wot, kC, kC);

  // 2) qkv projections into one (T x 2048) fp32 buffer
  auto gemm_blocks = [](int M, int N) { return (((M >> 4) * (N >> 6)) + 3) / 4; };
  gemm_bf16_wmma_kernel<<<gemm_blocks(kT, 1024), 128, 0, stream>>>(
      xb, wqt, qkv, kT, 1024, kC, 2048, 0);
  gemm_bf16_wmma_kernel<<<gemm_blocks(kT, 512), 128, 0, stream>>>(
      xb, wkt, qkv, kT, 512, kC, 2048, 1024);
  gemm_bf16_wmma_kernel<<<gemm_blocks(kT, 512), 128, 0, stream>>>(
      xb, wvt, qkv, kT, 512, kC, 2048, 1536);

  // 3) gate/ve add + rope + rms -> bf16 head-major q/k + key-major v^T
  qkv_epilogue_kernel<<<kT, 256, 0, stream>>>(qkv, x, ve, cosb, sinb, Wg, qb, kb, vt);

  // 4) fused attention (128 KB dynamic LDS for the 16 x 2048 score block)
  attention_kernel<<<dim3(kT / 16, kH), 256, 16 * kT * sizeof(float), stream>>>(
      qb, kb, vt, prev, cbias, convw, alpha, wout, yb);

  // 5) out = y @ Wo
  gemm_bf16_wmma_kernel<<<gemm_blocks(kT, kC), 128, 0, stream>>>(
      yb, wot, outp, kT, kC, kC, kC, 0);
}
